// GRU_47966194761795
// MI455X (gfx1250) — matrix operand order repaired and checked by reading, not compiled, not run on hardware
//
#include <hip/hip_runtime.h>

// ---------------------------------------------------------------------------
// MI455X (gfx1250) fused GRU: bf16 WMMA + Tensor Data Mover (TDM) staging.
//   B=4096, DX=512, DC=64 (DXC=576), H=2048 (3H=6144), OUT=512
// Compute-bound (~347 GFLOP vs ~260 MB traffic) -> v_wmma_f32_16x16x32_bf16.
// TDM DMAs double-buffered A/B tiles into LDS (pad_enable -> 144B row stride,
// bank-conflict-free ds_load_b128 fragment reads); one wave issues the D#s,
// s_wait_tensorcnt + workgroup barriers do the handshake; WMMA overlaps DMA.
// NOTE: the TDM builtin's LDS write is invisible to alias analysis (the LDS
// address is an integer in the descriptor), so we publish the LDS buffer via
// an empty asm with a memory clobber to keep the ds_load_b128 reads alive.
// ---------------------------------------------------------------------------

typedef __attribute__((ext_vector_type(16))) __bf16 v16bf;
typedef __attribute__((ext_vector_type(8)))  float  v8f;
typedef __attribute__((ext_vector_type(4)))  unsigned int u32x4;
typedef __attribute__((ext_vector_type(8)))  int          i32x8;
typedef __attribute__((ext_vector_type(4)))  int          i32x4;

#define B_SZ   4096
#define DX_SZ  512
#define DC_SZ  64
#define DXC    576
#define H_SZ   2048
#define H3     6144
#define OUT_SZ 512

#define KT           64                      // K elements staged per step
#define LDS_ROW      144                     // 64*2B + 16B pad (bank-conflict-free)
#define A_TILE_BYTES (128 * LDS_ROW)         // 18432
#define B_TILE_BYTES (64 * LDS_ROW)          // 9216
#define STAGE_BYTES  (A_TILE_BYTES + 3 * B_TILE_BYTES)  // 46080; x2 buffers = 90 KB

union Frag16 { uint4 u[2]; v16bf v; };

// Make the TDM's LDS writes visible to the compiler: escape the pointer into
// an asm statement that may read/write memory. Generates no instructions.
__device__ __forceinline__ void lds_publish(void* p) {
  asm volatile("" : "+v"(p) : : "memory");
}

__device__ __forceinline__ unsigned short f32_to_bf16_rne(float f) {
  union { float f; unsigned u; } v; v.f = f;
  unsigned r = v.u + 0x7FFFu + ((v.u >> 16) & 1u);
  return (unsigned short)(r >> 16);
}

__device__ __forceinline__ v8f wmma_bf16(v16bf a, v16bf b, v8f c) {
  return __builtin_amdgcn_wmma_f32_16x16x32_bf16(false, a, false, b, (short)0, c,
                                                 false, false);
}

// ---- TDM: 2D tile load Global -> LDS (D# per CDNA5 ISA ch.8) ---------------
// data_size=2B, pad every 32 dwords (128B row) by 4 dwords -> LDS stride 144B.
__device__ __forceinline__ void tdm_load_2d(const void* gaddr, unsigned lds_off,
                                            unsigned tensor_d0, unsigned row_stride,
                                            unsigned tile_d0, unsigned tile_d1) {
  const unsigned long long ga = (unsigned long long)gaddr;
  const unsigned tensor_d1 = 1u << 20;  // rows: large (tiles always in-bounds)
  u32x4 g0;
  g0[0] = 1u;                                          // count=1, user mode
  g0[1] = lds_off;                                     // lds_addr (bytes)
  g0[2] = (unsigned)(ga & 0xFFFFFFFFu);                // global_addr[31:0]
  g0[3] = (unsigned)((ga >> 32) & 0x01FFFFFFu) | (2u << 30);  // [56:32] | type=2
  i32x8 g1;
  g1[0] = (int)((1u << 16)        // data_size = 2 bytes
              | (1u << 20)        // pad_enable
              | (4u << 22)        // pad_interval code 4 -> every 32 dwords
              | (3u << 25));      // pad_amount  code 3 -> 4 dwords (16B)
  g1[1] = (int)((tensor_d0 & 0xFFFFu) << 16);                        // dim0[15:0]
  g1[2] = (int)(((tensor_d0 >> 16) & 0xFFFFu) | ((tensor_d1 & 0xFFFFu) << 16));
  g1[3] = (int)(((tensor_d1 >> 16) & 0xFFFFu) | ((tile_d0 & 0xFFFFu) << 16));
  g1[4] = (int)(tile_d1 & 0xFFFFu);                    // tile_dim1 | tile_dim2=0
  g1[5] = (int)row_stride;                             // tensor_dim0_stride lo32
  g1[6] = 0;                                           // stride hi | dim1_stride
  g1[7] = 0;
  const i32x4 gz  = {0, 0, 0, 0};
#if defined(__clang_major__) && (__clang_major__ >= 23)
  const i32x8 gz8 = {0, 0, 0, 0, 0, 0, 0, 0};
  __builtin_amdgcn_tensor_load_to_lds(g0, g1, gz, gz, gz8, 0);
#else
  __builtin_amdgcn_tensor_load_to_lds(g0, g1, gz, gz, 0);
#endif
}

// Stage one K-slab: A tile [128 x KT] + 3 gate B tiles [64 x KT] -> 4 TDM ops.
__device__ __forceinline__ void stage_tiles(const unsigned short* Ain,
                                            const unsigned short* W, int K,
                                            int m0blk, int n0blk, int kpos,
                                            unsigned lbase) {
  tdm_load_2d(Ain + (size_t)m0blk * K + kpos, lbase, (unsigned)K, (unsigned)K,
              KT, 128);
#pragma unroll
  for (int g = 0; g < 3; ++g)
    tdm_load_2d(W + (size_t)(g * H_SZ + n0blk) * K + kpos,
                lbase + A_TILE_BYTES + g * B_TILE_BYTES, (unsigned)K,
                (unsigned)K, KT, 64);
}

// ---- LDS fragment readers (rows at 144B stride) ----------------------------
// A frag 16x32 (MxK): lane L, m=(L&15), hi=L>>4: elems [kk+hi*8..+7],[+16..]
__device__ __forceinline__ v16bf lds_frag_a(const unsigned char* sm, int row,
                                            int kk, int lane) {
  const unsigned char* p = sm + row * LDS_ROW + kk * 2 + ((lane >> 4) << 4);
  Frag16 f;
  f.u[0] = *reinterpret_cast<const uint4*>(p);
  f.u[1] = *reinterpret_cast<const uint4*>(p + 32);
  return f.v;
}
// B frag 32x16 (KxN): lane L, n=(L&15): elems [kk+hi*8..+7],[+16..]
__device__ __forceinline__ v16bf lds_frag_b(const unsigned char* sm, int row,
                                            int kk, int lane) {
  const unsigned char* p = sm + row * LDS_ROW + kk * 2 + ((lane >> 4) << 4);
  Frag16 f;
  f.u[0] = *reinterpret_cast<const uint4*>(p);
  f.u[1] = *reinterpret_cast<const uint4*>(p + 32);
  return f.v;
}

// One GEMM phase: A[B,K] x W[3H,K]^T, TDM double-buffered through LDS.
// r->acc[0], z->acc[1] (both phases share), n->acc[NSET] (2=i_n, 3=h_n).
template <int NSET>
__device__ __forceinline__ void gru_phase_staged(
    const unsigned short* __restrict__ Ain, const unsigned short* __restrict__ W,
    int K, int m0blk, int n0blk, int mwave, int nwave, int lane, bool issuer,
    unsigned char* smem, v8f (&acc)[4][2][2]) {
  const int ml = lane & 15;
  if (issuer) stage_tiles(Ain, W, K, m0blk, n0blk, 0, 0u);
  int buf = 0;
  for (int kt = 0; kt < K; kt += KT) {
    const int nxt = kt + KT;
    if (issuer) {
      if (nxt < K) {
        stage_tiles(Ain, W, K, m0blk, n0blk, nxt,
                    (unsigned)((buf ^ 1) * STAGE_BYTES));
        __builtin_amdgcn_s_wait_tensorcnt(4);  // oldest 4 (current buf) done
      } else {
        __builtin_amdgcn_s_wait_tensorcnt(0);
      }
    }
    __syncthreads();
    lds_publish(smem);  // TDM wrote LDS: keep ds reads live & ordered
    const unsigned char* sA = smem + buf * STAGE_BYTES;
    const unsigned char* sB = sA + A_TILE_BYTES;
#pragma unroll
    for (int kk = 0; kk < KT; kk += 32) {
      v16bf a0 = lds_frag_a(sA, mwave + ml,      kk, lane);
      v16bf a1 = lds_frag_a(sA, mwave + 16 + ml, kk, lane);
#pragma unroll
      for (int g = 0; g < 3; ++g) {
        const int set = (g == 2) ? NSET : g;
        const unsigned char* sBg = sB + g * B_TILE_BYTES;
        v16bf b0 = lds_frag_b(sBg, nwave + ml,      kk, lane);
        v16bf b1 = lds_frag_b(sBg, nwave + 16 + ml, kk, lane);
        acc[set][0][0] = wmma_bf16(a0, b0, acc[set][0][0]);
        acc[set][0][1] = wmma_bf16(a0, b1, acc[set][0][1]);
        acc[set][1][0] = wmma_bf16(a1, b0, acc[set][1][0]);
        acc[set][1][1] = wmma_bf16(a1, b1, acc[set][1][1]);
      }
    }
    __syncthreads();  // all waves done with buf before it is re-staged
    buf ^= 1;
  }
}

// Fused GRU layer: workgroup tile 128x64 (8 waves, 32x32 each); gates stay in
// registers; h' = (1-z)*tanh(i_n + r*h_n) + z*h_prev.
__global__ __launch_bounds__(256) void gru_layer_wmma(
    const unsigned short* __restrict__ Xin, int Kx,          // [B, Kx] bf16
    const unsigned short* __restrict__ Wih,                  // [3H, Kx] bf16
    const unsigned short* __restrict__ Hprev,                // [B, H] bf16
    const unsigned short* __restrict__ Whh,                  // [3H, H] bf16
    const float* __restrict__ bih, const float* __restrict__ bhh,  // [3H]
    const float* __restrict__ HprevF,                        // [B, H] f32
    unsigned short* __restrict__ Hout) {                     // [B, H] bf16
  __shared__ __align__(16) unsigned char smem[2 * STAGE_BYTES];
  const int lane  = threadIdx.x & 31;
  const int wave  = threadIdx.x >> 5;
  const int waveM = wave & 3, waveN = wave >> 2;
  const int m0blk = blockIdx.y * 128, n0blk = blockIdx.x * 64;
  const int mwave = waveM * 32, nwave = waveN * 32;
  const bool issuer = (wave == 0);

  v8f acc[4][2][2] = {};  // 0:r(sum)  1:z(sum)  2:i_n  3:h_n

  gru_phase_staged<2>(Xin,   Wih, Kx,   m0blk, n0blk, mwave, nwave, lane,
                      issuer, smem, acc);
  gru_phase_staged<3>(Hprev, Whh, H_SZ, m0blk, n0blk, mwave, nwave, lane,
                      issuer, smem, acc);

  const int m0 = m0blk + mwave, n0 = n0blk + nwave;
  const int ml = lane & 15, hi = lane >> 4;
#pragma unroll
  for (int nf = 0; nf < 2; ++nf) {
    const int gn = n0 + nf * 16 + ml;
    const float bir = bih[0 * H_SZ + gn], bhr = bhh[0 * H_SZ + gn];
    const float biz = bih[1 * H_SZ + gn], bhz = bhh[1 * H_SZ + gn];
    const float bin = bih[2 * H_SZ + gn], bhn = bhh[2 * H_SZ + gn];
#pragma unroll
    for (int mf = 0; mf < 2; ++mf) {
#pragma unroll
      for (int r = 0; r < 8; ++r) {
        const int gm = m0 + mf * 16 + hi * 8 + r;
        const float xr = acc[0][mf][nf][r] + bir + bhr;
        const float xz = acc[1][mf][nf][r] + biz + bhz;
        const float rg = 1.0f / (1.0f + __expf(-xr));
        const float zg = 1.0f / (1.0f + __expf(-xz));
        const float ng = tanhf(acc[2][mf][nf][r] + bin +
                               rg * (acc[3][mf][nf][r] + bhn));
        const float hp = HprevF[(size_t)gm * H_SZ + gn];
        const float h  = (1.0f - zg) * ng + zg * hp;
        Hout[(size_t)gm * H_SZ + gn] = f32_to_bf16_rne(h);
      }
    }
  }
}

// ---- Decoder GEMM (2.5% of FLOPs): direct-load WMMA path -------------------
__device__ __forceinline__ v16bf load_frag_a_g(const unsigned short* rowk, int lane) {
  const unsigned short* p = rowk + ((lane >> 4) << 3);
  Frag16 f;
  f.u[0] = *reinterpret_cast<const uint4*>(p);
  f.u[1] = *reinterpret_cast<const uint4*>(p + 16);
  return f.v;
}
__device__ __forceinline__ v16bf load_frag_b_g(const unsigned short* rowk, int lane) {
  const unsigned short* p = rowk + ((lane >> 4) << 3);
  Frag16 f;
  f.u[0] = *reinterpret_cast<const uint4*>(p);
  f.u[1] = *reinterpret_cast<const uint4*>(p + 16);
  return f.v;
}

__global__ __launch_bounds__(256) void dec_gemm_wmma(
    const unsigned short* __restrict__ A,     // [B, H] bf16
    const unsigned short* __restrict__ W,     // [OUT, H] bf16
    const float* __restrict__ bias,
    float* __restrict__ Out) {
  const int lane  = threadIdx.x & 31;
  const int wave  = threadIdx.x >> 5;
  const int waveM = wave & 3, waveN = wave >> 2;
  const int m0 = blockIdx.y * 128 + waveM * 32;
  const int n0 = blockIdx.x * 64 + waveN * 32;
  const int ml = lane & 15;

  v8f acc[2][2] = {};
  for (int k = 0; k < H_SZ; k += 32) {
    v16bf a0 = load_frag_a_g(A + (size_t)(m0 +      ml) * H_SZ + k, lane);
    v16bf a1 = load_frag_a_g(A + (size_t)(m0 + 16 + ml) * H_SZ + k, lane);
    const unsigned short* wb = W + (size_t)(n0 + ml) * H_SZ + k;
    v16bf b0 = load_frag_b_g(wb, lane);
    v16bf b1 = load_frag_b_g(wb + (size_t)16 * H_SZ, lane);
    acc[0][0] = wmma_bf16(a0, b0, acc[0][0]);
    acc[0][1] = wmma_bf16(a0, b1, acc[0][1]);
    acc[1][0] = wmma_bf16(a1, b0, acc[1][0]);
    acc[1][1] = wmma_bf16(a1, b1, acc[1][1]);
  }
  const int hi = lane >> 4;
#pragma unroll
  for (int nf = 0; nf < 2; ++nf) {
    const int gn = n0 + nf * 16 + ml;
    const float bb = bias[gn];
#pragma unroll
    for (int mf = 0; mf < 2; ++mf)
#pragma unroll
      for (int r = 0; r < 8; ++r) {
        const int gm = m0 + mf * 16 + hi * 8 + r;
        Out[(size_t)gm * OUT_SZ + gn] = acc[mf][nf][r] + bb;
      }
  }
}

// ---- fp32 -> bf16 conversions ----------------------------------------------
__global__ void cvt_f32_to_bf16(const float* __restrict__ src,
                                unsigned short* __restrict__ dst, int n) {
  int i = blockIdx.x * blockDim.x + threadIdx.x;
  const int stride = gridDim.x * blockDim.x;
  for (; i < n; i += stride) dst[i] = f32_to_bf16_rne(src[i]);
}

__global__ void pack_xc_bf16(const float* __restrict__ x,
                             const float* __restrict__ c,
                             unsigned short* __restrict__ xc) {
  int i = blockIdx.x * blockDim.x + threadIdx.x;
  const int stride = gridDim.x * blockDim.x;
  const int n = B_SZ * DXC;
  for (; i < n; i += stride) {
    const int row = i / DXC, col = i - row * DXC;
    const float v = (col < DX_SZ) ? x[row * DX_SZ + col]
                                  : c[row * DC_SZ + (col - DX_SZ)];
    xc[i] = f32_to_bf16_rne(v);
  }
}

extern "C" void kernel_launch(void* const* d_in, const int* in_sizes, int n_in,
                              void* d_out, int out_size, void* d_ws, size_t ws_size,
                              hipStream_t stream) {
  const float* x     = (const float*)d_in[0];
  const float* c     = (const float*)d_in[1];
  const float* h0    = (const float*)d_in[2];   // [2, B, H]
  const float* w_ih0 = (const float*)d_in[3];
  const float* w_hh0 = (const float*)d_in[4];
  const float* b_ih0 = (const float*)d_in[5];
  const float* b_hh0 = (const float*)d_in[6];
  const float* w_ih1 = (const float*)d_in[7];
  const float* w_hh1 = (const float*)d_in[8];
  const float* b_ih1 = (const float*)d_in[9];
  const float* b_hh1 = (const float*)d_in[10];
  const float* w_dec = (const float*)d_in[11];
  const float* b_dec = (const float*)d_in[12];
  float* out = (float*)d_out;

  // bf16 workspace carve-out (~150 MB total)
  char* ws = (char*)d_ws;
  size_t off = 0;
  auto carve = [&](size_t elems) {
    unsigned short* p = (unsigned short*)(ws + off);
    off += ((elems * 2 + 255) / 256) * 256;
    return p;
  };
  unsigned short* xc_b   = carve((size_t)B_SZ * DXC);
  unsigned short* wih0_b = carve((size_t)H3 * DXC);
  unsigned short* whh0_b = carve((size_t)H3 * H_SZ);
  unsigned short* wih1_b = carve((size_t)H3 * H_SZ);
  unsigned short* whh1_b = carve((size_t)H3 * H_SZ);
  unsigned short* wdec_b = carve((size_t)OUT_SZ * H_SZ);
  unsigned short* h0l0_b = carve((size_t)B_SZ * H_SZ);
  unsigned short* h0l1_b = carve((size_t)B_SZ * H_SZ);
  unsigned short* h1_b   = carve((size_t)B_SZ * H_SZ);
  unsigned short* h2_b   = carve((size_t)B_SZ * H_SZ);

  const int T = 256;
  auto cgrid = [&](size_t n) {
    size_t g = (n + T - 1) / T;
    return (int)(g > 16384 ? 16384 : g);
  };

  pack_xc_bf16<<<cgrid((size_t)B_SZ * DXC), T, 0, stream>>>(x, c, xc_b);
  cvt_f32_to_bf16<<<cgrid((size_t)H3 * DXC), T, 0, stream>>>(w_ih0, wih0_b, H3 * DXC);
  cvt_f32_to_bf16<<<cgrid((size_t)H3 * H_SZ), T, 0, stream>>>(w_hh0, whh0_b, H3 * H_SZ);
  cvt_f32_to_bf16<<<cgrid((size_t)H3 * H_SZ), T, 0, stream>>>(w_ih1, wih1_b, H3 * H_SZ);
  cvt_f32_to_bf16<<<cgrid((size_t)H3 * H_SZ), T, 0, stream>>>(w_hh1, whh1_b, H3 * H_SZ);
  cvt_f32_to_bf16<<<cgrid((size_t)OUT_SZ * H_SZ), T, 0, stream>>>(w_dec, wdec_b, OUT_SZ * H_SZ);
  cvt_f32_to_bf16<<<cgrid((size_t)B_SZ * H_SZ), T, 0, stream>>>(h0, h0l0_b, B_SZ * H_SZ);
  cvt_f32_to_bf16<<<cgrid((size_t)B_SZ * H_SZ), T, 0, stream>>>(h0 + (size_t)B_SZ * H_SZ,
                                                                h0l1_b, B_SZ * H_SZ);

  dim3 blk(256);
  dim3 grdG(H_SZ / 64, B_SZ / 128);   // 32 x 32 workgroups
  gru_layer_wmma<<<grdG, blk, 0, stream>>>(xc_b, DXC, wih0_b, h0l0_b, whh0_b,
                                           b_ih0, b_hh0, h0, h1_b);
  gru_layer_wmma<<<grdG, blk, 0, stream>>>(h1_b, H_SZ, wih1_b, h0l1_b, whh1_b,
                                           b_ih1, b_hh1, h0 + (size_t)B_SZ * H_SZ,
                                           h2_b);
  dim3 grdD(OUT_SZ / 64, B_SZ / 128); // 8 x 32 workgroups
  dec_gemm_wmma<<<grdD, blk, 0, stream>>>(h2_b, wdec_b, b_dec, out);
}
